// HybridQNN_4767413699140
// MI455X (gfx1250) — matrix operand order repaired and checked by reading, not compiled, not run on hardware
//
#include <hip/hip_runtime.h>
#include <hip/hip_bf16.h>
#include <math.h>

typedef __attribute__((ext_vector_type(16))) _Float16 v16h;
typedef __attribute__((ext_vector_type(8)))  float    v8f;

struct cplx { float re, im; };
__device__ __forceinline__ cplx cmul(cplx a, cplx b) {
  return {a.re*b.re - a.im*b.im, a.re*b.im + a.im*b.re};
}
__device__ __forceinline__ cplx cadd(cplx a, cplx b) { return {a.re + b.re, a.im + b.im}; }

// ---------------------------------------------------------------------------
// Kernel 0: pack conv2 weights into WMMA B-fragment layout (once, 1 block).
// Fragment (nt,kb), lane l, element e  ->  n = 16*nt + (l&15),
//                                          k = e + 16*(l>>4) + 32*kb
// conv2_w flat index = oc*144 + k (k = ic*9 + ky*3 + kx); k>=144 zero-padded.
// ---------------------------------------------------------------------------
__global__ void __launch_bounds__(256) pack_w2_kernel(const float* __restrict__ w2,
                                                      _Float16* __restrict__ wp) {
  for (int idx = threadIdx.x; idx < 5120; idx += 256) {
    int e    = idx & 15;
    int lane = (idx >> 4) & 31;
    int fk   = idx >> 9;            // 0..9
    int kb   = fk % 5;
    int nt   = fk / 5;
    int oc   = nt * 16 + (lane & 15);
    int k    = e + 16 * (lane >> 4) + 32 * kb;
    wp[idx]  = (k < 144) ? (_Float16)w2[oc * 144 + k] : (_Float16)0.0f;
  }
}

// ---------------------------------------------------------------------------
// Kernel 1: conv1 (1->16, 3x3, pad1) + ReLU + 2x2 maxpool : 28x28 -> 16x14x14
// Zero-padded 30x30 input tile in LDS -> branch-free 36-MAC inner loop.
// ---------------------------------------------------------------------------
__global__ void __launch_bounds__(256) conv1_pool_kernel(
    const float* __restrict__ x, const float* __restrict__ w,
    const float* __restrict__ bias, _Float16* __restrict__ h1) {
  __shared__ float sIn[900];                       // 30x30 zero-padded
  __shared__ float sW[144];
  __shared__ float sB[16];
  const int b = blockIdx.x;
  const int tid = threadIdx.x;
  const float* xb = x + b * 784;
  for (int i = tid; i < 900; i += 256) {           // single-pass padded fill
    int yy = i / 30, xx = i - yy * 30;
    bool interior = (unsigned)(yy - 1) < 28u && (unsigned)(xx - 1) < 28u;
    sIn[i] = interior ? xb[(yy - 1) * 28 + (xx - 1)] : 0.0f;
  }
  for (int i = tid; i < 144; i += 256) sW[i] = w[i];
  if (tid < 16) sB[tid] = bias[tid];
  __syncthreads();
  for (int i = tid; i < 3136; i += 256) {          // 16*14*14 pooled outputs
    int oc = i / 196;
    int r  = i - oc * 196;
    int py = r / 14;
    int px = r - py * 14;
    const float* wk = &sW[oc * 9];
    float bv = sB[oc];
    float m = 0.0f;                                // relu>=0 -> max-with-0 == pool(relu)
    #pragma unroll
    for (int dy = 0; dy < 2; ++dy) {
      #pragma unroll
      for (int dx = 0; dx < 2; ++dx) {
        int base = (2 * py + dy) * 30 + (2 * px + dx); // padded coords
        float acc = bv;
        #pragma unroll
        for (int ky = 0; ky < 3; ++ky)
          #pragma unroll
          for (int kx = 0; kx < 3; ++kx)
            acc += sIn[base + ky * 30 + kx] * wk[ky * 3 + kx];
        m = fmaxf(m, acc);
      }
    }
    h1[b * 3136 + i] = (_Float16)m;
  }
}

// ---------------------------------------------------------------------------
// Kernel 2 (fused): conv2 WMMA implicit GEMM + bias/ReLU + pool + dense(->4)
//                   + 4-qubit circuit + post(4->10).  One WG / sample.
// GEMM: M=196 positions (13x16 tiles), N=32 oc (2x16), K=144 (5x32).
// sO padded to 208 rows so the D epilogue is fully branch-free.
// ---------------------------------------------------------------------------
__global__ void __launch_bounds__(256) conv2_qnn_kernel(
    const _Float16* __restrict__ h1, const _Float16* __restrict__ wp,
    const float* __restrict__ b2,
    const float* __restrict__ dw, const float* __restrict__ db,
    const float* __restrict__ qp, const float* __restrict__ pw,
    const float* __restrict__ pb, float* __restrict__ out) {
  __shared__ alignas(32) _Float16 sHp[18 * 256]; // padded acts [18ch][16][16]; ch16-17 = 0
  __shared__ alignas(32) _Float16 sA[13 * 5 * 32 * 16]; // im2col in A-fragment layout
  __shared__ alignas(32) _Float16 sWf[5120];     // B fragments (copied from wp)
  __shared__ float sO[208 * 32];                 // conv2 out [p=208 padded][oc=32]
  __shared__ float sB2[32];
  __shared__ float sRed[1024];                   // dense reduction [256][4]

  const int b   = blockIdx.x;
  const int tid = threadIdx.x;

  // ---- stage: padded activations (single pass, race-free) ----
  const _Float16* hb = h1 + b * 3136;
  for (int i = tid; i < 18 * 256; i += 256) {
    int ic = i >> 8, rem = i & 255, yy = rem >> 4, xx = rem & 15;
    bool interior = ic < 16 && (unsigned)(yy - 1) < 14u && (unsigned)(xx - 1) < 14u;
    sHp[i] = interior ? hb[ic * 196 + (yy - 1) * 14 + (xx - 1)] : (_Float16)0.0f;
  }
  // ---- stage: packed B fragments (straight 10 KB copy) ----
  for (int i = tid; i < 640; i += 256)
    ((uint4*)sWf)[i] = ((const uint4*)wp)[i];
  if (tid < 32) sB2[tid] = b2[tid];
  __syncthreads();

  // ---- build im2col A in fragment layout; k fully unrolled => all offsets
  //      are compile-time constants: 160 straight-line ds_load/ds_store pairs ----
  if (tid < 208) {
    int p  = tid;                       // A-matrix row (spatial position), 13 tiles
    int pc = p < 196 ? p : 195;         // clamp: rows >=196 are discarded later
    int y  = pc / 14, x = pc - (pc / 14) * 14;
    int gbase = y * 16 + x;             // padded-coord base (pad folds into ky,kx)
    int dbase = (p >> 4) * 2560 + (p & 15) * 16;
    #pragma unroll
    for (int k = 0; k < 160; ++k) {
      const int ic = k / 9, r9 = k % 9, ky = r9 / 3, kx = r9 % 3;   // consts
      const int kb = k >> 5, kl = k & 31;
      const int hf = kl >> 4, e = kl & 15;   // consts
      sA[dbase + kb * 512 + hf * 256 + e] = sHp[gbase + ic * 256 + ky * 16 + kx];
    }
  }
  __syncthreads();

  // ---- conv2 GEMM on v_wmma_f32_16x16x32_f16: pure b128 fragment loads ----
  const int lane = tid & 31;
  const int wave = tid >> 5;
  const int half = lane >> 4;
  const int lrow = lane & 15;
  const float bias0 = sB2[lrow];            // hoisted: depend only on lane
  const float bias1 = sB2[16 + lrow];
  for (int t = wave; t < 13; t += 8) {      // wave-uniform bound
    v8f acc0 = {}; v8f acc1 = {};
    #pragma unroll
    for (int kb = 0; kb < 5; ++kb) {
      v16h a  = *(const v16h*)&sA[(t * 5 + kb) * 512 + lane * 16];
      v16h b0 = *(const v16h*)&sWf[(kb * 32 + lane) * 16];
      v16h b1 = *(const v16h*)&sWf[((5 + kb) * 32 + lane) * 16];
      acc0 = __builtin_amdgcn_wmma_f32_16x16x32_f16(false, a, false, b0,
                                                    (short)0, acc0, false, false);
      acc1 = __builtin_amdgcn_wmma_f32_16x16x32_f16(false, a, false, b1,
                                                    (short)0, acc1, false, false);
    }
    // D layout: lane -> n, vgpr r -> m = r + 8*half. Branch-free (padded sO).
    int pbase = (t * 16 + 8 * half) * 32 + lrow;
    #pragma unroll
    for (int r = 0; r < 8; ++r) {
      sO[pbase + r * 32]      = fmaxf(acc0[r] + bias0, 0.0f);
      sO[pbase + r * 32 + 16] = fmaxf(acc1[r] + bias1, 0.0f);
    }
  }
  __syncthreads();

  // ---- 2x2 maxpool + dense (1568 -> 4) partials ----
  float part[4] = {0.f, 0.f, 0.f, 0.f};
  for (int f = tid; f < 1568; f += 256) {          // f = c*49 + py*7 + px
    int c  = f / 49;
    int r  = f - c * 49;
    int py = r / 7;
    int px = r - py * 7;
    int p00 = (2 * py) * 14 + 2 * px;
    float v = fmaxf(fmaxf(sO[p00 * 32 + c],        sO[(p00 + 1) * 32 + c]),
                    fmaxf(sO[(p00 + 14) * 32 + c], sO[(p00 + 15) * 32 + c]));
    #pragma unroll
    for (int j = 0; j < 4; ++j) part[j] += v * dw[j * 1568 + f];
  }
  #pragma unroll
  for (int j = 0; j < 4; ++j) sRed[tid * 4 + j] = part[j];
  __syncthreads();
  for (int s = 128; s > 0; s >>= 1) {
    if (tid < s) {
      #pragma unroll
      for (int j = 0; j < 4; ++j) sRed[tid * 4 + j] += sRed[(tid + s) * 4 + j];
    }
    __syncthreads();
  }

  // ---- 4-qubit statevector circuit + post head (tiny; one lane) ----
  if (tid == 0) {
    const float PI = 3.14159265358979323846f;
    float ang[4];
    #pragma unroll
    for (int j = 0; j < 4; ++j) ang[j] = sRed[j] + db[j];

    float ar[16], ai[16];
    #pragma unroll
    for (int i = 0; i < 16; ++i) { ar[i] = 0.f; ai[i] = 0.f; }
    ar[0] = 1.f;

    // Initial real rotations: M=[[c,-s],[s,c]], theta = ang*pi, half-angle
    for (int w = 0; w < 4; ++w) {
      float th = ang[w] * PI * 0.5f;
      float c = cosf(th), s = sinf(th);
      int bit = 8 >> w;
      for (int i0 = 0; i0 < 16; ++i0) {
        if (i0 & bit) continue;
        int i1 = i0 | bit;
        float a0r = ar[i0], a0i = ai[i0], a1r = ar[i1], a1i = ai[i1];
        ar[i0] = c * a0r - s * a1r;  ai[i0] = c * a0i - s * a1i;
        ar[i1] = s * a0r + c * a1r;  ai[i1] = s * a0i + c * a1i;
      }
    }
    // Layers: Rz(g)Ry(b)Rx(a) per qubit, then CNOT+Rz entanglers
    for (int l = 0; l < 2; ++l) {
      int base = l * 18;
      for (int q = 0; q < 4; ++q) {
        float ha = qp[base + q * 3 + 0] * 0.5f;
        float hy = qp[base + q * 3 + 1] * 0.5f;
        float hg = qp[base + q * 3 + 2] * 0.5f;
        float ca = cosf(ha), sa = sinf(ha);
        float cb = cosf(hy), sb = sinf(hy);
        float cg = cosf(hg), sg = sinf(hg);
        cplx t00 = { cb * ca,  sb * sa};           // T = Ry @ Rx
        cplx t01 = {-sb * ca, -cb * sa};
        cplx t10 = { sb * ca, -cb * sa};
        cplx t11 = { cb * ca, -sb * sa};
        cplx ep = {cg, -sg}, em = {cg, sg};        // e^{-ig/2}, e^{+ig/2}
        cplx m00 = cmul(ep, t00), m01 = cmul(ep, t01);
        cplx m10 = cmul(em, t10), m11 = cmul(em, t11);
        int bit = 8 >> q;
        for (int i0 = 0; i0 < 16; ++i0) {
          if (i0 & bit) continue;
          int i1 = i0 | bit;
          cplx a0 = {ar[i0], ai[i0]}, a1 = {ar[i1], ai[i1]};
          cplx n0 = cadd(cmul(m00, a0), cmul(m01, a1));
          cplx n1 = cadd(cmul(m10, a0), cmul(m11, a1));
          ar[i0] = n0.re; ai[i0] = n0.im;
          ar[i1] = n1.re; ai[i1] = n1.im;
        }
      }
      // kron(I2,Rz(t)) @ CNOT = diag(e,e*,e,e*).CNOT -> phase + target swap
      int kidx = 0;
      for (int i = 0; i < 4; ++i) {
        for (int j = i + 1; j < 4; ++j, ++kidx) {
          float th = qp[base + 12 + kidx] * 0.5f;
          cplx e  = {cosf(th), -sinf(th)};
          cplx ec = {e.re, -e.im};
          int bc = 8 >> i, bt = 8 >> j;
          for (int bb = 0; bb < 16; ++bb) {
            if (bb & (bc | bt)) continue;
            int i00 = bb, i01 = bb | bt, i10 = bb | bc, i11 = bb | bc | bt;
            cplx a00 = {ar[i00], ai[i00]}, a01 = {ar[i01], ai[i01]};
            cplx a10 = {ar[i10], ai[i10]}, a11 = {ar[i11], ai[i11]};
            cplx n00 = cmul(e, a00),  n01 = cmul(ec, a01);
            cplx n10 = cmul(e, a11),  n11 = cmul(ec, a10);
            ar[i00] = n00.re; ai[i00] = n00.im;
            ar[i01] = n01.re; ai[i01] = n01.im;
            ar[i10] = n10.re; ai[i10] = n10.im;
            ar[i11] = n11.re; ai[i11] = n11.im;
          }
        }
      }
    }
    float z[4];
    for (int w = 0; w < 4; ++w) {
      int bit = 8 >> w;
      float acc = 0.f;
      for (int i = 0; i < 16; ++i) {
        float p = ar[i] * ar[i] + ai[i] * ai[i];
        acc += (i & bit) ? -p : p;
      }
      z[w] = acc;
    }
    float* ob = out + b * 10;
    for (int o = 0; o < 10; ++o) {
      float r = pb[o];
      #pragma unroll
      for (int w = 0; w < 4; ++w) r += z[w] * pw[o * 4 + w];
      ob[o] = r;
    }
  }
}

extern "C" void kernel_launch(void* const* d_in, const int* in_sizes, int n_in,
                              void* d_out, int out_size, void* d_ws, size_t ws_size,
                              hipStream_t stream) {
  (void)n_in; (void)out_size; (void)ws_size;
  const float* x   = (const float*)d_in[0];
  const float* c1w = (const float*)d_in[1];
  const float* c1b = (const float*)d_in[2];
  const float* c2w = (const float*)d_in[3];
  const float* c2b = (const float*)d_in[4];
  const float* dw  = (const float*)d_in[5];
  const float* db  = (const float*)d_in[6];
  const float* qp  = (const float*)d_in[7];
  const float* pw  = (const float*)d_in[8];
  const float* pb  = (const float*)d_in[9];
  float* out = (float*)d_out;

  const int B = in_sizes[0] / 784;                       // 4096
  _Float16* h1 = (_Float16*)d_ws;                        // B*3136 f16 = 25.7 MB
  _Float16* wpack = h1 + (size_t)B * 3136;               // 5120 f16, 16B-aligned

  pack_w2_kernel<<<1, 256, 0, stream>>>(c2w, wpack);
  conv1_pool_kernel<<<B, 256, 0, stream>>>(x, c1w, c1b, h1);
  conv2_qnn_kernel<<<B, 256, 0, stream>>>(h1, wpack, c2b, dw, db, qp, pw, pb, out);
}
